// FourierLinear_23270132810145
// MI455X (gfx1250) — compile-verified, matrix operand order checked by reading, not run on hardware
//
#include <hip/hip_runtime.h>
#include <hip/hip_bf16.h>

// out = x @ real(ifft2(sparse spectrum))*256, factorized into two f16 WMMA GEMMs.
//   A1[m,k]  = k<2048 ?  s[k]*cos(2pi*m*idx_in[k]/4096)  : -s[k-2048]*sin(...)
//   B1t[n,k] = k<2048 ?       cos(2pi*n*idx_out[k]/4096) :           sin(...)
//   Wt[n,m]  = (1/65536) * (A1 @ B1t^T)[m,n]   (stage 1, f16 transposed store)
//   out      = Xh @ Wt^T                        (stage 2, f32 store)

typedef __attribute__((ext_vector_type(16))) _Float16 v16h;
typedef __attribute__((ext_vector_type(8)))  _Float16 v8h;
typedef __attribute__((ext_vector_type(4)))  _Float16 v4h;
typedef __attribute__((ext_vector_type(8)))  float    v8f;
typedef __attribute__((ext_vector_type(4)))  int      v4i;

union HFrag { v16h v; v8h h[2]; };

#define IN_F   4096
#define OUT_F  4096
#define NFREQ  2048
#define MROWS  8192
#define BM     256
#define BN     128
#define BK     64
#define PAD_ROW 72            // 64 f16 + 8 pad -> 144B stride (9*16B aligned, conflict-free)

// ---- gfx1250 async global->LDS (probe-verified live on this toolchain) ----
#if defined(__has_builtin)
# if __has_builtin(__builtin_amdgcn_global_load_async_to_lds_b128)
#  define HAVE_ASYNC_LDS 1
# endif
#endif
#ifndef HAVE_ASYNC_LDS
# define HAVE_ASYNC_LDS 0
#endif

__device__ __forceinline__ void wait_async_zero() {
#if defined(__has_builtin) && __has_builtin(__builtin_amdgcn_s_wait_asynccnt)
    __builtin_amdgcn_s_wait_asynccnt(0);
#else
    asm volatile("s_wait_asynccnt 0x0" ::: "memory");
#endif
}

#if HAVE_ASYNC_LDS
typedef __attribute__((address_space(1))) v4i* gptr_v4i;
typedef __attribute__((address_space(3))) v4i* lptr_v4i;
__device__ __forceinline__ void async_copy_b128(const _Float16* g, _Float16* l) {
    __builtin_amdgcn_global_load_async_to_lds_b128(
        (gptr_v4i)(uintptr_t)g,
        (lptr_v4i)(uint32_t)(uintptr_t)l,     // low 32 bits of generic ptr = LDS offset
        0, 0);
}
#endif

// ---------------------------------------------------------------- convert x -> f16
__global__ __launch_bounds__(256) void convert_x_kernel(const float* __restrict__ x,
                                                        _Float16* __restrict__ xh, int n) {
    int i = (blockIdx.x * blockDim.x + threadIdx.x) * 4;
    if (i < n) {
        float4 v = *reinterpret_cast<const float4*>(&x[i]);
        v4h h = { (_Float16)v.x, (_Float16)v.y, (_Float16)v.z, (_Float16)v.w };
        *reinterpret_cast<v4h*>(&xh[i]) = h;
    }
}

// ---------------------------------------------------------------- trig factor tables
__global__ __launch_bounds__(256) void build_tables_kernel(const float* __restrict__ spec,
                                                           const int* __restrict__ idx_in,
                                                           const int* __restrict__ idx_out,
                                                           _Float16* __restrict__ A1,
                                                           _Float16* __restrict__ B1t) {
    int t = blockIdx.x * blockDim.x + threadIdx.x;   // 0 .. 4096*4096-1
    int k = t & (IN_F - 1);
    int r = t >> 12;                                  // m for A1, n for B1t
    int f = k & (NFREQ - 1);
    bool isSin = k >= NFREQ;
    int ki = idx_in[f];
    int ko = idx_out[f];
    float s = spec[f];
    const float W = 6.2831853071795864769f / 4096.0f;
    float sa, ca, sb, cb;
    __sincosf((float)((r * ki) & 4095) * W, &sa, &ca);
    __sincosf((float)((r * ko) & 4095) * W, &sb, &cb);
    A1 [(size_t)r * IN_F + k] = (_Float16)(isSin ? -s * sa : s * ca);
    B1t[(size_t)r * IN_F + k] = (_Float16)(isSin ?      sb :     cb);
}

// ---------------------------------------------------------------- WMMA GEMM
// C[M,N] = A[M,K] * Bt[N,K]^T.  MODE 0: f32 row-major store (*scale)
//                               MODE 1: f16 transposed store Ct[n*M+m] (*scale)
// 256x128 block tile, 8 waves with 64x64 wave tiles (16 WMMA : 16 LDS-loads per
// K-step), BK=64 slabs, double-buffered LDS, async global->LDS pipeline.
template <int MODE>
__global__ __launch_bounds__(256)
void wmma_gemm_kernel(const _Float16* __restrict__ A, const _Float16* __restrict__ Bt,
                      float* __restrict__ Cf, _Float16* __restrict__ Ct,
                      int M, int N, int K, float scale) {
    __shared__ _Float16 ldsA[2][BM * PAD_ROW];       // 2 x 36 KB
    __shared__ _Float16 ldsB[2][BN * PAD_ROW];       // 2 x 18 KB

    const int tid  = threadIdx.x;
    const int lane = tid & 31;                 // wave32
    const int wave = tid >> 5;                 // 8 waves
    const int wm   = wave & 3;                 // 0..3 : 64-row strip
    const int wn   = wave >> 2;                // 0..1 : 64-col strip
    const int m0   = blockIdx.y * BM;
    const int n0   = blockIdx.x * BN;
    const int lrow  = lane & 15;
    const int halfk = (lane >> 4) * 8;

    // Tile loads (16B chunks, 8 f16 each): A 256x64 -> 2048 chunks (8/thread),
    //                                      B 128x64 -> 1024 chunks (4/thread).
    int lofsA[8], lofsB[4];
    const _Float16* gA[8];
    const _Float16* gB[4];
#pragma unroll
    for (int i = 0; i < 8; ++i) {
        int c   = tid + 256 * i;
        int row = c >> 3;                      // 0..255
        int col = (c & 7) << 3;                // 0..56
        lofsA[i] = row * PAD_ROW + col;
        gA[i]    = &A[(size_t)(m0 + row) * K + col];
    }
#pragma unroll
    for (int i = 0; i < 4; ++i) {
        int c   = tid + 256 * i;
        int row = c >> 3;                      // 0..127
        int col = (c & 7) << 3;
        lofsB[i] = row * PAD_ROW + col;
        gB[i]    = &Bt[(size_t)(n0 + row) * K + col];
    }

    v8f acc[4][4];
#pragma unroll
    for (int ti = 0; ti < 4; ++ti)
#pragma unroll
        for (int tj = 0; tj < 4; ++tj)
#pragma unroll
            for (int e = 0; e < 8; ++e) acc[ti][tj][e] = 0.0f;

    const int nk = K / BK;

    // ---- prologue: slab 0 -> buffer 0
#if HAVE_ASYNC_LDS
#pragma unroll
    for (int i = 0; i < 8; ++i) async_copy_b128(gA[i], &ldsA[0][lofsA[i]]);
#pragma unroll
    for (int i = 0; i < 4; ++i) async_copy_b128(gB[i], &ldsB[0][lofsB[i]]);
    wait_async_zero();
#else
    {
        int4 ra[8], rb[4];
#pragma unroll
        for (int i = 0; i < 8; ++i) ra[i] = *reinterpret_cast<const int4*>(gA[i]);
#pragma unroll
        for (int i = 0; i < 4; ++i) rb[i] = *reinterpret_cast<const int4*>(gB[i]);
#pragma unroll
        for (int i = 0; i < 8; ++i) *reinterpret_cast<int4*>(&ldsA[0][lofsA[i]]) = ra[i];
#pragma unroll
        for (int i = 0; i < 4; ++i) *reinterpret_cast<int4*>(&ldsB[0][lofsB[i]]) = rb[i];
    }
#endif
    __syncthreads();

    for (int kt = 0; kt < nk; ++kt) {
        const int buf = kt & 1;
        const bool hasNext = (kt + 1) < nk;
        const int off = (kt + 1) * BK;         // next slab K offset (f16 elements)

        // ---- issue next slab's async copies early (hidden behind 32 WMMAs)
#if HAVE_ASYNC_LDS
        if (hasNext) {
#pragma unroll
            for (int i = 0; i < 8; ++i)
                async_copy_b128(gA[i] + off, &ldsA[buf ^ 1][lofsA[i]]);
#pragma unroll
            for (int i = 0; i < 4; ++i)
                async_copy_b128(gB[i] + off, &ldsB[buf ^ 1][lofsB[i]]);
        }
#else
        int4 na[8], nb[4];
        if (hasNext) {
#pragma unroll
            for (int i = 0; i < 8; ++i) na[i] = *reinterpret_cast<const int4*>(gA[i] + off);
#pragma unroll
            for (int i = 0; i < 4; ++i) nb[i] = *reinterpret_cast<const int4*>(gB[i] + off);
        }
#endif

        // ---- compute current slab: 2 K-steps of 32, 16 WMMA each
#pragma unroll
        for (int ks = 0; ks < BK; ks += 32) {
            HFrag af[4], bf[4];
#pragma unroll
            for (int ti = 0; ti < 4; ++ti) {
                int r = (wm * 64 + ti * 16 + lrow) * PAD_ROW + ks;
                af[ti].h[0] = *reinterpret_cast<const v8h*>(&ldsA[buf][r + halfk]);
                af[ti].h[1] = *reinterpret_cast<const v8h*>(&ldsA[buf][r + 16 + halfk]);
            }
#pragma unroll
            for (int tj = 0; tj < 4; ++tj) {
                int r = (wn * 64 + tj * 16 + lrow) * PAD_ROW + ks;
                bf[tj].h[0] = *reinterpret_cast<const v8h*>(&ldsB[buf][r + halfk]);
                bf[tj].h[1] = *reinterpret_cast<const v8h*>(&ldsB[buf][r + 16 + halfk]);
            }
#pragma unroll
            for (int ti = 0; ti < 4; ++ti)
#pragma unroll
                for (int tj = 0; tj < 4; ++tj)
                    acc[ti][tj] = __builtin_amdgcn_wmma_f32_16x16x32_f16(
                        false, af[ti].v, false, bf[tj].v,
                        (short)0, acc[ti][tj], false, false);
        }

        // ---- commit next slab to the other buffer
#if HAVE_ASYNC_LDS
        if (hasNext) wait_async_zero();
#else
        if (hasNext) {
#pragma unroll
            for (int i = 0; i < 8; ++i) *reinterpret_cast<int4*>(&ldsA[buf ^ 1][lofsA[i]]) = na[i];
#pragma unroll
            for (int i = 0; i < 4; ++i) *reinterpret_cast<int4*>(&ldsB[buf ^ 1][lofsB[i]]) = nb[i];
        }
#endif
        __syncthreads();
    }

    // ---- epilogue (32-bit flat indices; max 8192*4096 fits u32).
    // f32 C layout: VGPR r, lanes 0-15 -> M=r, lanes 16-31 -> M=8+r, N=lane&15.
    const int mo = (lane >> 4) * 8;
#pragma unroll
    for (int ti = 0; ti < 4; ++ti) {
#pragma unroll
        for (int tj = 0; tj < 4; ++tj) {
            int mbase = m0 + wm * 64 + ti * 16 + mo;
            int nn    = n0 + wn * 64 + tj * 16 + lrow;
            if (MODE == 0) {
#pragma unroll
                for (int r = 0; r < 8; ++r) {
                    uint32_t o = (uint32_t)(mbase + r) * (uint32_t)N + (uint32_t)nn;
                    Cf[o] = acc[ti][tj][r] * scale;
                }
            } else {
                v8h h;
#pragma unroll
                for (int r = 0; r < 8; ++r) h[r] = (_Float16)(acc[ti][tj][r] * scale);
                uint32_t o = (uint32_t)nn * (uint32_t)M + (uint32_t)mbase;
                *reinterpret_cast<v8h*>(&Ct[o]) = h;   // b128 transposed store
            }
        }
    }
}

// ---------------------------------------------------------------- launch
extern "C" void kernel_launch(void* const* d_in, const int* in_sizes, int n_in,
                              void* d_out, int out_size, void* d_ws, size_t ws_size,
                              hipStream_t stream) {
    const float* x        = (const float*)d_in[0];   // [4,2048,4096] f32
    const float* spectrum = (const float*)d_in[1];   // [2048] f32
    const int*   indices  = (const int*)d_in[2];     // [2,2048] i32
    const int*   idx_in   = indices;
    const int*   idx_out  = indices + NFREQ;
    float*       out      = (float*)d_out;           // [8192,4096] f32

    char* ws = (char*)d_ws;
    _Float16* Xh  = (_Float16*)(ws);                                   //  64 MB
    _Float16* A1  = (_Float16*)(ws + (size_t)64  * 1024 * 1024);       //  32 MB
    _Float16* B1t = (_Float16*)(ws + (size_t)96  * 1024 * 1024);       //  32 MB
    _Float16* Wt  = (_Float16*)(ws + (size_t)128 * 1024 * 1024);       //  32 MB

    const int NX = MROWS * IN_F;                                       // 33,554,432
    convert_x_kernel<<<NX / (256 * 4), 256, 0, stream>>>(x, Xh, NX);

    build_tables_kernel<<<(IN_F * IN_F) / 256, 256, 0, stream>>>(
        spectrum, idx_in, idx_out, A1, B1t);

    dim3 blk(256);
    dim3 g1(OUT_F / BN, IN_F / BM);                  // (32, 16)
    wmma_gemm_kernel<1><<<g1, blk, 0, stream>>>(A1, B1t, nullptr, Wt,
                                                IN_F, OUT_F, IN_F, 1.0f / 65536.0f);
    dim3 g2(OUT_F / BN, MROWS / BM);                 // (32, 32)
    wmma_gemm_kernel<0><<<g2, blk, 0, stream>>>(Xh, Wt, out, nullptr,
                                                MROWS, OUT_F, IN_F, 1.0f);
}
